// BinaryLinear_19473381720569
// MI455X (gfx1250) — 8-bit matrix operand order repaired and checked by reading, compiled, not run on hardware
//
#include <hip/hip_runtime.h>

typedef int v8i __attribute__((ext_vector_type(8)));

namespace {
constexpr int kInF     = 256;
constexpr int kOutF    = 256;
constexpr int kBM      = 128;
constexpr int kThreads = 256;
}

__device__ __forceinline__ int pack_sign4(float4 v) {
  int s0 = (v.x > 0.0f) - (v.x < 0.0f);
  int s1 = (v.y > 0.0f) - (v.y < 0.0f);
  int s2 = (v.z > 0.0f) - (v.z < 0.0f);
  int s3 = (v.w > 0.0f) - (v.w < 0.0f);
  return (s0 & 255) | ((s1 & 255) << 8) | ((s2 & 255) << 16) | ((s3 & 255) << 24);
}

__global__ __launch_bounds__(kThreads)
void binlinear_wmma_iu8(const float* __restrict__ x,
                        const float* __restrict__ w,
                        float* __restrict__ out) {
  __shared__ int ldsW[16384];

  const int tid = threadIdx.x;

  const float4* w4 = (const float4*)w;
  #pragma unroll 4
  for (int i = 0; i < 64; ++i) {
    int idx = i * kThreads + tid;
    float4 v = w4[idx];
    int o     = idx >> 6;
    int kb    = (idx & 63) << 2;
    int ct    = o >> 4;
    int n     = o & 15;
    int kc    = kb >> 6;
    int k64   = kb & 63;
    int g     = k64 >> 5;
    int khalf = (k64 >> 4) & 1;
    int byteoff = k64 & 15;
    int addr = (((ct * 4 + kc) * 2 + g) << 9) + ((khalf * 16 + n) << 4) + byteoff;
    ldsW[addr >> 2] = pack_sign4(v);
  }
  __syncthreads();

  const int wave = tid >> 5;
  const int lane = tid & 31;
  const int n    = lane & 15;
  const int h    = lane >> 4;

  const size_t mbase = (size_t)blockIdx.x * kBM + (size_t)wave * 16;
  const float* xrow  = x + (mbase + n) * kInF;

  v8i a[4];
  #pragma unroll
  for (int kc = 0; kc < 4; ++kc) {
    const int base = kc * 64 + h * 16;
    a[kc][0] = pack_sign4(*(const float4*)(xrow + base + 0));
    a[kc][1] = pack_sign4(*(const float4*)(xrow + base + 4));
    a[kc][2] = pack_sign4(*(const float4*)(xrow + base + 8));
    a[kc][3] = pack_sign4(*(const float4*)(xrow + base + 12));
    a[kc][4] = pack_sign4(*(const float4*)(xrow + base + 32));
    a[kc][5] = pack_sign4(*(const float4*)(xrow + base + 36));
    a[kc][6] = pack_sign4(*(const float4*)(xrow + base + 40));
    a[kc][7] = pack_sign4(*(const float4*)(xrow + base + 44));
  }

  for (int ct = 0; ct < 16; ++ct) {
    v8i c = {0, 0, 0, 0, 0, 0, 0, 0};
    #pragma unroll
    for (int kc = 0; kc < 4; ++kc) {
      const char* p = (const char*)ldsW + (((ct * 4 + kc) * 2) << 9) + lane * 16;
      int4 lo = *(const int4*)(p);
      int4 hi = *(const int4*)(p + 512);
      v8i b;
      b[0] = lo.x; b[1] = lo.y; b[2] = lo.z; b[3] = lo.w;
      b[4] = hi.x; b[5] = hi.y; b[6] = hi.z; b[7] = hi.w;
      c = __builtin_amdgcn_wmma_i32_16x16x64_iu8(true, a[kc], true, b, c,
                                                 false, false);
    }
    float* obase = out + (mbase + (size_t)h * 8) * kOutF + ct * 16 + n;
    #pragma unroll
    for (int r = 0; r < 8; ++r) {
      obase[(size_t)r * kOutF] = (float)c[r];
    }
  }
}

extern "C" void kernel_launch(void* const* d_in, const int* in_sizes, int n_in,
                              void* d_out, int out_size, void* d_ws, size_t ws_size,
                              hipStream_t stream) {
  (void)n_in; (void)out_size; (void)d_ws; (void)ws_size;
  const float* x = (const float*)d_in[0];
  const float* w = (const float*)d_in[1];
  float* out = (float*)d_out;
  const int n_rows = in_sizes[0] / kInF;
  dim3 grid(n_rows / kBM), block(kThreads);
  binlinear_wmma_iu8<<<grid, block, 0, stream>>>(x, w, out);
}
